// VLSTMModel_13967233647327
// MI455X (gfx1250) — compile-verified, matrix operand order checked by reading, not run on hardware
//
#include <hip/hip_runtime.h>
#include <hip/hip_bf16.h>

typedef __attribute__((ext_vector_type(16))) _Float16 v16h;
typedef __attribute__((ext_vector_type(8)))  _Float16 v8h;
typedef __attribute__((ext_vector_type(8)))  float    v8f;

union Frag16 { v16h v; v8h h2[2]; };

#define TT   20
#define NN   100000
#define INX  2
#define EMB  64
#define RNN  128
#define OUTD 5
#define G4   512          // 4*RNN
#define ROWS_PER_BLOCK 128
#define THREADS 256

// ---- LDS layout (bytes) ----
#define OFF_BIH   0                        // 64 frags * 512 halfs  = 65536 B
#define OFF_BHH   (OFF_BIH + 65536)        // 128 frags * 512 halfs = 131072 B
#define OFF_BOUT  (OFF_BHH + 131072)       // 4 frags * 512 halfs   = 4096 B
#define OFF_HC    (OFF_BOUT + 4096)        // carry h  f16, 8*16*128 = 32768 B
#define OFF_HNT   (OFF_HC + 32768)         // h_new    f16, 8*16*128 = 32768 B
#define OFF_WEMB  (OFF_HNT + 32768)        // 2*64 f32              = 512 B
#define OFF_BEMB  (OFF_WEMB + 512)         // 64 f32                = 256 B
#define OFF_BIAS  (OFF_BEMB + 256)         // 512 f32               = 2048 B
#define LDS_BYTES (OFF_BIAS + 2048)        // = 269056 B  (< 320 KB)

// K-slot inside a 16x16x32 f16 A/B fragment for element e (0..15) of a lane.
// VGPR v holds K-pairs: v<4 -> K=2v,2v+1 ; v>=4 -> K=2v+8,2v+9 ; hi half-wave +8.
// Per lane this is two contiguous 8-half runs: [8hi .. 8hi+7] and [16+8hi .. 16+8hi+7].
__device__ __forceinline__ int frag_k(int e, int hi) {
    int v = e >> 1;
    return 2 * v + ((v >= 4) ? 8 : 0) + 8 * hi + (e & 1);
}

__device__ __forceinline__ float sigm(float x) {
    return 1.0f / (1.0f + __expf(-x));
}

__device__ __forceinline__ float tanh_fast(float x) {
#if __has_builtin(__builtin_amdgcn_tanhf)
    return __builtin_amdgcn_tanhf(x);          // v_tanh_f32 on gfx1250
#else
    return 2.0f * sigm(2.0f * x) - 1.0f;
#endif
}

__device__ __forceinline__ v8f wmma_f16(v16h a, v16h b, v8f c) {
    return __builtin_amdgcn_wmma_f32_16x16x32_f16(false, a, false, b,
                                                  (short)0, c, false, false);
}

// load one A fragment (16x32 f16) from an f16 row-major [16][128] tile: 2x b128
__device__ __forceinline__ v16h load_afrag(const _Float16* tile, int nl, int hi, int kc) {
    Frag16 f;
    const _Float16* p = tile + nl * RNN + 8 * hi + 32 * kc;
    f.h2[0] = *(const v8h*)(p);
    f.h2[1] = *(const v8h*)(p + 16);
    return f.v;
}

__global__ __launch_bounds__(THREADS, 2)   // cap VGPRs: 2 waves/SIMD
void vlstm_kernel(const float* __restrict__ input_data,   // (T,N,IN)
                  const float* __restrict__ hidden0,      // (N,RNN)
                  const float* __restrict__ cell0,        // (N,RNN)
                  const int*   __restrict__ mask,         // (T,N)
                  const float* __restrict__ W_emb,        // (IN,EMB)
                  const float* __restrict__ b_emb,        // (EMB)
                  const float* __restrict__ W_ih,         // (4R,EMB)
                  const float* __restrict__ b_ih,         // (4R)
                  const float* __restrict__ W_hh,         // (4R,RNN)
                  const float* __restrict__ b_hh,         // (4R)
                  const float* __restrict__ W_out,        // (OUT,RNN)
                  const float* __restrict__ b_out,        // (OUT)
                  float* __restrict__ out,                // (T,N,OUT)
                  float* __restrict__ hfin,               // (N,RNN)
                  float* __restrict__ cfin)               // (N,RNN)
{
    extern __shared__ char smem[];
    _Float16* sBih  = (_Float16*)(smem + OFF_BIH);
    _Float16* sBhh  = (_Float16*)(smem + OFF_BHH);
    _Float16* sBout = (_Float16*)(smem + OFF_BOUT);
    _Float16* sHC   = (_Float16*)(smem + OFF_HC);    // carried h, f16, [wave][16][128]
    _Float16* sHNT  = (_Float16*)(smem + OFF_HNT);   // h_new,     f16, [wave][16][128]
    float*    sWemb = (float*)   (smem + OFF_WEMB);
    float*    sBemb = (float*)   (smem + OFF_BEMB);
    float*    sBias = (float*)   (smem + OFF_BIAS);

    const int tid   = threadIdx.x;
    const int lane  = tid & 31;
    const int wave  = tid >> 5;
    const int hi    = lane >> 4;
    const int nl    = lane & 15;
    const int blockRow = blockIdx.x * ROWS_PER_BLOCK;
    const int waveRow  = blockRow + wave * 16;

    // ---------- one-time staging into LDS ----------
    for (int i = tid; i < 64 * 512; i += THREADS) {
        int f  = i >> 9;
        int le = i & 511;
        int l  = le >> 4, e = le & 15;
        int kc = f >> 5,  nt = f & 31;
        int k  = 32 * kc + frag_k(e, l >> 4);
        int col = 16 * nt + (l & 15);
        sBih[i] = (_Float16)W_ih[col * EMB + k];
    }
    for (int i = tid; i < 128 * 512; i += THREADS) {
        int f  = i >> 9;
        int le = i & 511;
        int l  = le >> 4, e = le & 15;
        int kc = f >> 5,  nt = f & 31;
        int k  = 32 * kc + frag_k(e, l >> 4);
        int col = 16 * nt + (l & 15);
        sBhh[i] = (_Float16)W_hh[col * RNN + k];
    }
    for (int i = tid; i < 4 * 512; i += THREADS) {
        int f  = i >> 9;
        int le = i & 511;
        int l  = le >> 4, e = le & 15;
        int k  = 32 * f + frag_k(e, l >> 4);
        int col = l & 15;
        sBout[i] = (col < OUTD) ? (_Float16)W_out[col * RNN + k] : (_Float16)0.0f;
    }
    for (int i = tid; i < INX * EMB; i += THREADS) sWemb[i] = W_emb[i];
    for (int i = tid; i < EMB;       i += THREADS) sBemb[i] = b_emb[i];
    for (int i = tid; i < G4;        i += THREADS) sBias[i] = b_ih[i] + b_hh[i];

    for (int i = tid; i < 8 * 16 * RNN; i += THREADS) {
        int w = i >> 11;
        int r = (i >> 7) & 15;
        int cc = i & 127;
        int row = blockRow + w * 16 + r;
        int rowc = row < NN ? row : NN - 1;
        sHC[i] = (_Float16)hidden0[rowc * RNN + cc];
    }

    // c state in registers, D layout: creg[j][rr] = c[row = rr + 8*hi][col = j*16 + nl]
    float creg[8][8];
    {
        int rbase = waveRow + 8 * hi;
        #pragma unroll
        for (int j = 0; j < 8; ++j) {
            #pragma unroll
            for (int rr = 0; rr < 8; ++rr) {
                int row = rbase + rr;
                int rowc = row < NN ? row : NN - 1;
                creg[j][rr] = cell0[rowc * RNN + j * 16 + nl];
            }
        }
    }
    const float bo = (nl < OUTD) ? b_out[nl] : 0.0f;

    __syncthreads();

    _Float16* myHC  = sHC  + wave * (16 * RNN);
    _Float16* myHNT = sHNT + wave * (16 * RNN);

    // ---------- time loop ----------
    for (int t = 0; t < TT; ++t) {
        // emb A fragments (row M = nl), built scalar since K(IN)=2
        const int myRow  = waveRow + nl;
        const int myRowC = myRow < NN ? myRow : NN - 1;
        const float x0 = input_data[(size_t)t * NN * INX + myRowC * INX + 0];
        const float x1 = input_data[(size_t)t * NN * INX + myRowC * INX + 1];
        v16h aemb[2];
        #pragma unroll
        for (int kc = 0; kc < 2; ++kc) {
            #pragma unroll
            for (int e = 0; e < 16; ++e) {
                int k = 32 * kc + frag_k(e, hi);
                float v = fmaf(x0, sWemb[k], fmaf(x1, sWemb[EMB + k], sBemb[k]));
                aemb[kc][e] = (_Float16)fmaxf(v, 0.0f);
            }
        }
        // h A fragments: pure packed ds_load_b128 pairs, no conversion
        v16h ah[4];
        #pragma unroll
        for (int kc = 0; kc < 4; ++kc) ah[kc] = load_afrag(myHC, nl, hi, kc);

        // per-row mask bits for D layout (rows rr + 8*hi), packed in one register
        int mbits = 0;
        {
            int rbase = waveRow + 8 * hi;
            #pragma unroll
            for (int rr = 0; rr < 8; ++rr) {
                int row = rbase + rr;
                int m = (row < NN) ? mask[(size_t)t * NN + row] : 0;
                mbits |= (m ? 1 : 0) << rr;
            }
        }

        #pragma unroll
        for (int j = 0; j < 8; ++j) {
            // 4 interleaved gate chains, B fragments double-buffered per K-group.
            v8f acc0 = {}, acc1 = {}, acc2 = {}, acc3 = {};
            v16h bcur[4], bnxt[4];
            #pragma unroll
            for (int g = 0; g < 4; ++g)            // preload K-group 0 (W_ih kc=0)
                bcur[g] = *(const v16h*)(sBih + ((g * 8 + j) * 512) + lane * 16);
            #pragma unroll
            for (int kc = 0; kc < 6; ++kc) {
                if (kc < 5) {                       // prefetch next K-group
                    int kn = kc + 1;
                    const _Float16* base = (kn < 2) ? (sBih + kn * 32 * 512)
                                                    : (sBhh + (kn - 2) * 32 * 512);
                    #pragma unroll
                    for (int g = 0; g < 4; ++g)
                        bnxt[g] = *(const v16h*)(base + ((g * 8 + j) * 512) + lane * 16);
                }
                v16h a = (kc < 2) ? aemb[kc] : ah[kc - 2];
                acc0 = wmma_f16(a, bcur[0], acc0);
                acc1 = wmma_f16(a, bcur[1], acc1);
                acc2 = wmma_f16(a, bcur[2], acc2);
                acc3 = wmma_f16(a, bcur[3], acc3);
                #pragma unroll
                for (int g = 0; g < 4; ++g) bcur[g] = bnxt[g];   // renamed, free
            }
            // bias + pointwise LSTM cell; h_new stashed unconditionally (f16)
            const float bi = sBias[(0 * 8 + j) * 16 + nl];
            const float bf = sBias[(1 * 8 + j) * 16 + nl];
            const float bg = sBias[(2 * 8 + j) * 16 + nl];
            const float bb = sBias[(3 * 8 + j) * 16 + nl];
            #pragma unroll
            for (int rr = 0; rr < 8; ++rr) {
                float iv = sigm(acc0[rr] + bi);
                float fv = sigm(acc1[rr] + bf);
                float gv = tanh_fast(acc2[rr] + bg);
                float ov = sigm(acc3[rr] + bb);
                float cn = fv * creg[j][rr] + iv * gv;
                float hn = ov * tanh_fast(cn);
                if (mbits & (1 << rr)) creg[j][rr] = cn;
                int ridx = rr + 8 * hi;
                int col  = j * 16 + nl;
                myHNT[ridx * RNN + col] = (_Float16)hn;
            }
        }
        __syncthreads();   // sHNT complete across lanes

        // output head: out = h_new @ W_out^T + b_out   (1 n-tile, 4 K-chunks)
        v8f oacc = {};
        #pragma unroll
        for (int kc = 0; kc < 4; ++kc) {
            v16h ahn = load_afrag(myHNT, nl, hi, kc);
            v16h b   = *(const v16h*)(sBout + kc * 512 + lane * 16);
            oacc = wmma_f16(ahn, b, oacc);
        }
        if (nl < OUTD) {
            int rbase = waveRow + 8 * hi;
            #pragma unroll
            for (int rr = 0; rr < 8; ++rr) {
                int row = rbase + rr;
                if (row < NN) {
                    float v = (mbits & (1 << rr)) ? (oacc[rr] + bo) : 0.0f;
                    out[(size_t)t * NN * OUTD + (size_t)row * OUTD + nl] = v;
                }
            }
        }

        // packed masked carry: sHC = mask(row) ? h_new : sHC, 16B chunks (1 row each)
        #pragma unroll
        for (int c2 = 0; c2 < 8; ++c2) {
            int idx  = lane + 32 * c2;      // 256 chunks of 8 halfs per wave tile
            int row  = idx >> 4;            // 16 chunks per row
            int grow = waveRow + row;
            int m = (grow < NN) ? mask[(size_t)t * NN + grow] : 0;
            if (m) *(v8h*)(myHC + idx * 8) = *(const v8h*)(myHNT + idx * 8);
        }
        __syncthreads();   // sHC stable before next step's fragment reads
    }

    // ---------- final states ----------
    {
        int rbase = waveRow + 8 * hi;
        #pragma unroll
        for (int j = 0; j < 8; ++j) {
            #pragma unroll
            for (int rr = 0; rr < 8; ++rr) {
                int row = rbase + rr;
                if (row < NN) cfin[(size_t)row * RNN + j * 16 + nl] = creg[j][rr];
            }
        }
    }
    for (int i = tid; i < 8 * 16 * RNN; i += THREADS) {
        int w = i >> 11;
        int r = (i >> 7) & 15;
        int cc = i & 127;
        int row = blockRow + w * 16 + r;
        if (row < NN) hfin[(size_t)row * RNN + cc] = (float)sHC[i];
    }
}

extern "C" void kernel_launch(void* const* d_in, const int* in_sizes, int n_in,
                              void* d_out, int out_size, void* d_ws, size_t ws_size,
                              hipStream_t stream) {
    const float* input_data = (const float*)d_in[0];
    const float* hidden0    = (const float*)d_in[1];
    const float* cell0      = (const float*)d_in[2];
    const int*   mask       = (const int*)  d_in[3];
    const float* W_emb      = (const float*)d_in[4];
    const float* b_emb      = (const float*)d_in[5];
    const float* W_ih       = (const float*)d_in[6];
    const float* b_ih       = (const float*)d_in[7];
    const float* W_hh       = (const float*)d_in[8];
    const float* b_hh       = (const float*)d_in[9];
    const float* W_out      = (const float*)d_in[10];
    const float* b_out      = (const float*)d_in[11];

    float* out  = (float*)d_out;                       // (T,N,OUT)
    float* hfin = out + (size_t)TT * NN * OUTD;        // (N,RNN)
    float* cfin = hfin + (size_t)NN * RNN;             // (N,RNN)

    (void)hipFuncSetAttribute((const void*)vlstm_kernel,
                              hipFuncAttributeMaxDynamicSharedMemorySize,
                              LDS_BYTES);

    int blocks = (NN + ROWS_PER_BLOCK - 1) / ROWS_PER_BLOCK;
    vlstm_kernel<<<blocks, THREADS, LDS_BYTES, stream>>>(
        input_data, hidden0, cell0, mask,
        W_emb, b_emb, W_ih, b_ih, W_hh, b_hh, W_out, b_out,
        out, hfin, cfin);
}
